// MoCA_91336774517529
// MI455X (gfx1250) — matrix operand order repaired and checked by reading, not compiled, not run on hardware
//
#include <hip/hip_runtime.h>

typedef __attribute__((ext_vector_type(16))) _Float16 v16h;
typedef __attribute__((ext_vector_type(8)))  _Float16 v8h;
typedef __attribute__((ext_vector_type(8)))  float    v8f;

#define HW   4096
#define CCH  512
#define LCH  64
#define PCH  256
#define NB   8

// ---------------------------------------------------------------------------
// WMMA helpers (gfx1250 wave32, V_WMMA_F32_16X16X32_F16)
// ---------------------------------------------------------------------------
__device__ inline v8f wmma_f16(v16h a, v16h b, v8f c) {
  return __builtin_amdgcn_wmma_f32_16x16x32_f16(
      /*neg_a=*/false, a, /*neg_b=*/false, b,
      /*c_mod=*/(short)0, c, /*reuse_a=*/false, /*reuse_b=*/false);
}

// A fragment 16x32 (MxK) from row-major [M,K] (row m contiguous in K).
// Per ISA 7.12.2: lane m=lane&15, hi=lane>>4; elems 0..7 -> K=hi*8+j,
// elems 8..15 -> K=16+hi*8+(j-8).
__device__ inline v16h load_A(const _Float16* base, int ld) {
  const int lane = threadIdx.x & 31;
  const int m = lane & 15, hi = lane >> 4;
  const _Float16* p = base + (size_t)m * ld;
  v8h lo = *(const v8h*)(p + hi * 8);
  v8h hh = *(const v8h*)(p + 16 + hi * 8);
  v16h r;
#pragma unroll
  for (int j = 0; j < 8; ++j) { r[j] = lo[j]; r[8 + j] = hh[j]; }
  return r;
}

// B fragment 32x16 (KxN) given B^T stored row-major [N,K] (row n contiguous K).
// Per ISA: lane n=lane&15, hi=lane>>4; elems 0..7 -> K=hi*8+j, elems 8..15 -> K=16+hi*8+(j-8).
__device__ inline v16h load_Bt(const _Float16* baseT, int ld) {
  const int lane = threadIdx.x & 31;
  const int n = lane & 15, hi = lane >> 4;
  const _Float16* p = baseT + (size_t)n * ld + hi * 8;
  v8h a = *(const v8h*)(p);
  v8h b = *(const v8h*)(p + 16);
  v16h r;
#pragma unroll
  for (int j = 0; j < 8; ++j) { r[j] = a[j]; r[8 + j] = b[j]; }
  return r;
}

// Store 16x16 f32 D tile as f16, row-major with leading dim ld.
// C/D layout: elem r, lane: N=lane&15, M=r+8*(lane>>4).
__device__ inline void store_tile_f16(_Float16* out, int ld, const v8f& d) {
  const int lane = threadIdx.x & 31;
  const int n = lane & 15, hi = lane >> 4;
#pragma unroll
  for (int r = 0; r < 8; ++r) out[(size_t)(r + 8 * hi) * ld + n] = (_Float16)d[r];
}

// ---------------------------------------------------------------------------
// Kernel 1: convert weights/concepts to f16 (gains folded in)
// ---------------------------------------------------------------------------
__global__ void prep_kernel(const float* __restrict__ wth, const float* __restrict__ wph,
                            const float* __restrict__ wg, const float* __restrict__ wo,
                            const float* __restrict__ concepts,
                            _Float16* __restrict__ Whth, _Float16* __restrict__ Whph,
                            _Float16* __restrict__ Whg, _Float16* __restrict__ Who,
                            _Float16* __restrict__ Ch, _Float16* __restrict__ Ct) {
  const float ginC = 0.04419417382415922f;  // 1/sqrt(512)
  const float ginL = 0.125f;                // 1/sqrt(64)
  const int stride = gridDim.x * blockDim.x;
  for (int i = blockIdx.x * blockDim.x + threadIdx.x; i < LCH * CCH; i += stride) {
    Whth[i] = (_Float16)(wth[i] * ginC);
    Whph[i] = (_Float16)(wph[i] * ginC);
    Whg[i]  = (_Float16)(wg[i]  * ginC);
    Who[i]  = (_Float16)(wo[i]  * ginL);   // w_o is [C,L], C*L == L*C elements
  }
  for (int i = blockIdx.x * blockDim.x + threadIdx.x; i < PCH * LCH; i += stride) {
    float v = concepts[i];                 // concepts: [P, L] row-major
    Ch[i] = (_Float16)v;                   // K matrix: [P keys, L]
    int p = i >> 6, l = i & 63;
    Ct[l * PCH + p] = (_Float16)v;         // V^T: [L, P]
  }
}

// ---------------------------------------------------------------------------
// Kernel 2: per-batch LDS-tiled transpose [R, Ccols] f32 -> [Ccols, R] f16
// ---------------------------------------------------------------------------
__global__ __launch_bounds__(256) void transpose_f32f16_kernel(
    const float* __restrict__ in, _Float16* __restrict__ out, int R, int Ccols) {
  __shared__ float tile[32][33];
  const int b = blockIdx.z;
  const float* src = in + (size_t)b * R * Ccols;
  _Float16* dst = out + (size_t)b * R * Ccols;
  const int c0 = blockIdx.x * 32, r0 = blockIdx.y * 32;
  const int tx = threadIdx.x, ty = threadIdx.y;
#pragma unroll
  for (int i = 0; i < 4; ++i)
    tile[ty + i * 8][tx] = src[(size_t)(r0 + ty + i * 8) * Ccols + c0 + tx];
  __syncthreads();
#pragma unroll
  for (int i = 0; i < 4; ++i)
    dst[(size_t)(c0 + ty + i * 8) * R + r0 + tx] = (_Float16)tile[tx][ty + i * 8];
}

__global__ __launch_bounds__(256) void transpose_f16_kernel(
    const _Float16* __restrict__ in, _Float16* __restrict__ out, int R, int Ccols) {
  __shared__ float tile[32][33];
  const int b = blockIdx.z;
  const _Float16* src = in + (size_t)b * R * Ccols;
  _Float16* dst = out + (size_t)b * R * Ccols;
  const int c0 = blockIdx.x * 32, r0 = blockIdx.y * 32;
  const int tx = threadIdx.x, ty = threadIdx.y;
#pragma unroll
  for (int i = 0; i < 4; ++i)
    tile[ty + i * 8][tx] = (float)src[(size_t)(r0 + ty + i * 8) * Ccols + c0 + tx];
  __syncthreads();
#pragma unroll
  for (int i = 0; i < 4; ++i)
    dst[(size_t)(c0 + ty + i * 8) * R + r0 + tx] = (_Float16)tile[tx][ty + i * 8];
}

// ---------------------------------------------------------------------------
// Kernel 3a: dual projection producing transposed outputs [HW, L]:
//   out1[hw,l] = sum_c X[hw,c] * W1[l,c] ;  out2 likewise with W2.
// One 16x16 tile per wave; A-fragment of X shared between both outputs.
// ---------------------------------------------------------------------------
__global__ __launch_bounds__(128) void projT2_kernel(
    const _Float16* __restrict__ X,
    const _Float16* __restrict__ W1, _Float16* __restrict__ out1,
    const _Float16* __restrict__ W2, _Float16* __restrict__ out2) {
  const int wave = threadIdx.x >> 5;
  const int gid = blockIdx.x * 4 + wave;        // 8192 waves
  const int b = gid >> 10;                      // 1024 tiles/batch
  const int rem = gid & 1023;
  const int m0 = (rem >> 2) * 16;               // 256 m-tiles (HW)
  const int n0 = (rem & 3) * 16;                // 4 n-tiles (L)
  const _Float16* Xb = X + (size_t)b * HW * CCH + (size_t)m0 * CCH;
  v8f acc1 = {}; v8f acc2 = {};
#pragma unroll 4
  for (int k = 0; k < CCH; k += 32) {
    v16h a = load_A(Xb + k, CCH);
    acc1 = wmma_f16(a, load_Bt(W1 + (size_t)n0 * CCH + k, CCH), acc1);
    acc2 = wmma_f16(a, load_Bt(W2 + (size_t)n0 * CCH + k, CCH), acc2);
  }
  store_tile_f16(out1 + (size_t)b * HW * LCH + (size_t)m0 * LCH + n0, LCH, acc1);
  store_tile_f16(out2 + (size_t)b * HW * LCH + (size_t)m0 * LCH + n0, LCH, acc2);
}

// Kernel 3b: single-output variant (enc projection), no branch in the K loop.
__global__ __launch_bounds__(128) void projT1_kernel(
    const _Float16* __restrict__ X,
    const _Float16* __restrict__ W1, _Float16* __restrict__ out1) {
  const int wave = threadIdx.x >> 5;
  const int gid = blockIdx.x * 4 + wave;
  const int b = gid >> 10;
  const int rem = gid & 1023;
  const int m0 = (rem >> 2) * 16;
  const int n0 = (rem & 3) * 16;
  const _Float16* Xb = X + (size_t)b * HW * CCH + (size_t)m0 * CCH;
  v8f acc1 = {};
#pragma unroll 4
  for (int k = 0; k < CCH; k += 32) {
    v16h a = load_A(Xb + k, CCH);
    acc1 = wmma_f16(a, load_Bt(W1 + (size_t)n0 * CCH + k, CCH), acc1);
  }
  store_tile_f16(out1 + (size_t)b * HW * LCH + (size_t)m0 * LCH + n0, LCH, acc1);
}

// ---------------------------------------------------------------------------
// Kernel 4: g projection in natural layout [L, HW]:
//   g[l,hw] = sum_c Wg[l,c] * X[hw,c]
// ---------------------------------------------------------------------------
__global__ __launch_bounds__(128) void projG_kernel(
    const _Float16* __restrict__ X, const _Float16* __restrict__ Wg,
    _Float16* __restrict__ g) {
  const int wave = threadIdx.x >> 5;
  const int gid = blockIdx.x * 4 + wave;        // 8192 waves
  const int b = gid >> 10;
  const int rem = gid & 1023;
  const int m0 = (rem >> 8) * 16;               // 4 m-tiles (L)
  const int n0 = (rem & 255) * 16;              // 256 n-tiles (HW)
  const _Float16* Xb = X + (size_t)b * HW * CCH + (size_t)n0 * CCH;
  v8f acc = {};
#pragma unroll 4
  for (int k = 0; k < CCH; k += 32) {
    v16h a = load_A(Wg + (size_t)m0 * CCH + k, CCH);
    acc = wmma_f16(a, load_Bt(Xb + k, CCH), acc);
  }
  store_tile_f16(g + (size_t)b * LCH * HW + (size_t)m0 * HW + n0, HW, acc);
}

// ---------------------------------------------------------------------------
// Kernel 5: fused flash attention (no masking; softmax over all Sk keys).
//   qT: [Sq,64] f16, kT: [Sk,64] f16, vT: [64,Sk] f16 -> out: [Sq,64] f16
// One wave = 16 query rows; key blocks of 32; online softmax; P relaid
// C-layout -> A-layout through per-wave LDS.
// ---------------------------------------------------------------------------
__global__ __launch_bounds__(256) void flash_kernel(
    const _Float16* __restrict__ qT, long qBS,
    const _Float16* __restrict__ kT, long kBS,
    const _Float16* __restrict__ vT, long vBS,
    _Float16* __restrict__ out, long oBS,
    int Sk, int qtPerBatch) {
  __shared__ _Float16 ldsP[8][16 * 32];
  const int wave = threadIdx.x >> 5;
  const int lane = threadIdx.x & 31;
  const int n = lane & 15, hi = lane >> 4;
  const int gid = blockIdx.x * 8 + wave;
  const int b = gid / qtPerBatch;
  const int q0 = (gid - b * qtPerBatch) * 16;
  const _Float16* Q = qT + (size_t)b * qBS + (size_t)q0 * LCH;
  const _Float16* K = kT + (size_t)b * kBS;
  const _Float16* V = vT + (size_t)b * vBS;
  _Float16* Pt = &ldsP[wave][0];

  const v16h qa0 = load_A(Q, LCH);
  const v16h qa1 = load_A(Q + 32, LCH);
  v8f o0 = {}, o1 = {}, o2 = {}, o3 = {};
  float mrow[8], lrow[8];
#pragma unroll
  for (int r = 0; r < 8; ++r) { mrow[r] = -3.0e38f; lrow[r] = 0.f; }

  for (int kb = 0; kb < Sk; kb += 32) {
    if (kb + 32 < Sk) __builtin_prefetch(K + (size_t)(kb + 32) * LCH, 0, 0);
    // S = Q @ K^T for keys [kb, kb+32): two 16x16 tiles
    v8f s0 = {}, s1 = {};
    s0 = wmma_f16(qa0, load_Bt(K + (size_t)kb * LCH, LCH), s0);
    s0 = wmma_f16(qa1, load_Bt(K + (size_t)kb * LCH + 32, LCH), s0);
    s1 = wmma_f16(qa0, load_Bt(K + (size_t)(kb + 16) * LCH, LCH), s1);
    s1 = wmma_f16(qa1, load_Bt(K + (size_t)(kb + 16) * LCH + 32, LCH), s1);
    // online softmax per row (rows live across 16-lane halves)
    float scale[8];
#pragma unroll
    for (int r = 0; r < 8; ++r) {
      float t = fmaxf(s0[r], s1[r]);
#pragma unroll
      for (int off = 1; off < 16; off <<= 1) t = fmaxf(t, __shfl_xor(t, off, 32));
      const float nm = fmaxf(mrow[r], t);
      scale[r] = __expf(mrow[r] - nm);
      mrow[r] = nm;
      const float p0 = __expf(s0[r] - nm);
      const float p1 = __expf(s1[r] - nm);
      float rs = p0 + p1;
#pragma unroll
      for (int off = 1; off < 16; off <<= 1) rs += __shfl_xor(rs, off, 32);
      lrow[r] = lrow[r] * scale[r] + rs;
      const int m = r + 8 * hi;
      Pt[m * 32 + n] = (_Float16)p0;        // keys kb..kb+15
      Pt[m * 32 + 16 + n] = (_Float16)p1;   // keys kb+16..kb+31
    }
#pragma unroll
    for (int r = 0; r < 8; ++r) {
      o0[r] *= scale[r]; o1[r] *= scale[r]; o2[r] *= scale[r]; o3[r] *= scale[r];
    }
    // O += P @ V   (V^T rows are d-indices, contiguous in key)
    const v16h pa = load_A(Pt, 32);
    o0 = wmma_f16(pa, load_Bt(V + (size_t)0  * Sk + kb, Sk), o0);
    o1 = wmma_f16(pa, load_Bt(V + (size_t)16 * Sk + kb, Sk), o1);
    o2 = wmma_f16(pa, load_Bt(V + (size_t)32 * Sk + kb, Sk), o2);
    o3 = wmma_f16(pa, load_Bt(V + (size_t)48 * Sk + kb, Sk), o3);
  }
  _Float16* O = out + (size_t)b * oBS + (size_t)q0 * LCH;
#pragma unroll
  for (int r = 0; r < 8; ++r) {
    const float inv = 1.0f / lrow[r];
    const int m = r + 8 * hi;
    O[m * LCH + 0  + n] = (_Float16)(o0[r] * inv);
    O[m * LCH + 16 + n] = (_Float16)(o1[r] * inv);
    O[m * LCH + 32 + n] = (_Float16)(o2[r] * inv);
    O[m * LCH + 48 + n] = (_Float16)(o3[r] * inv);
  }
}

// ---------------------------------------------------------------------------
// Kernel 6: output projection + residual:
//   dest[c,hw] = gamma * sum_l Who[c,l]*latT[hw,l] + residual[c,hw]
// (residual may alias dest for the in-place final pass — no __restrict__)
// ---------------------------------------------------------------------------
__global__ __launch_bounds__(128) void outproj_kernel(
    const _Float16* latT, const _Float16* Who, const float* residual,
    float* dest, const float* gammaPtr) {
  const int wave = threadIdx.x >> 5;
  const int lane = threadIdx.x & 31;
  const int gid = blockIdx.x * 4 + wave;        // 65536 waves
  const int b = gid >> 13;                      // 8192 tiles/batch
  const int rem = gid & 8191;
  const int m0 = (rem >> 8) * 16;               // 32 m-tiles (C)
  const int n0 = (rem & 255) * 16;              // 256 n-tiles (HW)
  const float gamma = *gammaPtr;
  const _Float16* Lb = latT + (size_t)b * HW * LCH + (size_t)n0 * LCH;
  v8f acc = {};
  acc = wmma_f16(load_A(Who + (size_t)m0 * LCH, LCH), load_Bt(Lb, LCH), acc);
  acc = wmma_f16(load_A(Who + (size_t)m0 * LCH + 32, LCH), load_Bt(Lb + 32, LCH), acc);
  const int n = lane & 15, hi = lane >> 4;
  const size_t base = (size_t)b * CCH * HW + (size_t)m0 * HW + n0 + n;
#pragma unroll
  for (int r = 0; r < 8; ++r) {
    const size_t idx = base + (size_t)(r + 8 * hi) * HW;
    dest[idx] = gamma * acc[r] + residual[idx];
  }
}

// ---------------------------------------------------------------------------
// Host-side orchestration
// ---------------------------------------------------------------------------
extern "C" void kernel_launch(void* const* d_in, const int* in_sizes, int n_in,
                              void* d_out, int out_size, void* d_ws, size_t ws_size,
                              hipStream_t stream) {
  const float* fm       = (const float*)d_in[0];  // [B, C, HW]
  const float* concepts = (const float*)d_in[1];  // [P, L]
  const float* w_theta  = (const float*)d_in[2];  // [L, C]
  const float* w_phi    = (const float*)d_in[3];
  const float* w_g      = (const float*)d_in[4];
  const float* w_o      = (const float*)d_in[5];  // [C, L]
  const float* gamma_sa = (const float*)d_in[6];
  const float* gamma_mo = (const float*)d_in[7];
  float* out = (float*)d_out;

  char* ws = (char*)d_ws;
  const size_t SZ_XH = (size_t)NB * HW * CCH * 2;   // 32 MB (Xh, later saH)
  const size_t SZ_A  = (size_t)NB * HW * LCH * 2;   // 4 MB each
  _Float16* Xh   = (_Float16*)(ws);                       // Xh / saH
  _Float16* thT  = (_Float16*)(ws + SZ_XH);               // thT / encT
  _Float16* phT  = (_Float16*)(ws + SZ_XH + SZ_A);        // phT / O2
  _Float16* gB   = (_Float16*)(ws + SZ_XH + 2 * SZ_A);    // g [L, HW]
  _Float16* O1   = (_Float16*)(ws + SZ_XH + 3 * SZ_A);    // attn1 out [HW, L]
  _Float16* latT = (_Float16*)(ws + SZ_XH + 4 * SZ_A);    // latT1 / latT2
  _Float16* WhTh = (_Float16*)(ws + SZ_XH + 5 * SZ_A);
  _Float16* WhPh = WhTh + LCH * CCH;
  _Float16* WhG  = WhPh + LCH * CCH;
  _Float16* WhO  = WhG + LCH * CCH;
  _Float16* Ch   = WhO + CCH * LCH;                       // concepts [P, L] f16
  _Float16* Ct   = Ch + PCH * LCH;                        // concepts^T [L, P] f16

  const long QS = (long)HW * LCH;

  // 1. weights + concepts -> f16 (gains folded)
  prep_kernel<<<64, 256, 0, stream>>>(w_theta, w_phi, w_g, w_o, concepts,
                                      WhTh, WhPh, WhG, WhO, Ch, Ct);
  // 2. feature_map [C,HW] -> Xh [HW,C] f16
  transpose_f32f16_kernel<<<dim3(HW / 32, CCH / 32, NB), dim3(32, 8), 0, stream>>>(
      fm, Xh, CCH, HW);
  // 3. theta^T, phi^T  [HW, L]
  projT2_kernel<<<2048, 128, 0, stream>>>(Xh, WhTh, thT, WhPh, phT);
  // 4. g [L, HW]
  projG_kernel<<<2048, 128, 0, stream>>>(Xh, WhG, gB);
  // 5. self-attention (flash, Sk = 4096)
  flash_kernel<<<256, 256, 0, stream>>>(thT, QS, phT, QS, gB, QS, O1, QS,
                                        HW, HW / 16);
  // 6. raw-reinterpret [L,HW] view of O1 -> latT [HW, L]
  transpose_f16_kernel<<<dim3(HW / 32, LCH / 32, NB), dim3(32, 8), 0, stream>>>(
      O1, latT, LCH, HW);
  // 7. sa_out = gamma_sa * w_o(lat) + feature_map   (written into d_out)
  outproj_kernel<<<16384, 128, 0, stream>>>(latT, WhO, fm, out, gamma_sa);
  // 8. sa_out [C,HW] -> saH [HW,C] f16 (reuse Xh region)
  transpose_f32f16_kernel<<<dim3(HW / 32, CCH / 32, NB), dim3(32, 8), 0, stream>>>(
      out, Xh, CCH, HW);
  // 9. enc^T [HW, L] (reuse thT region)
  projT1_kernel<<<2048, 128, 0, stream>>>(Xh, WhTh, thT);
  // 10. MoCA attention: K/V = concepts (batch stride 0), Sk = 256 -> O2 (phT region)
  flash_kernel<<<256, 256, 0, stream>>>(thT, QS, Ch, 0, Ct, 0, phT, QS,
                                        PCH, HW / 16);
  // 11. reinterpret+transpose O2 -> latT
  transpose_f16_kernel<<<dim3(HW / 32, LCH / 32, NB), dim3(32, 8), 0, stream>>>(
      phT, latT, LCH, HW);
  // 12. out = gamma_moca * w_o(lat2) + sa_out   (in-place on d_out)
  outproj_kernel<<<16384, 128, 0, stream>>>(latT, WhO, out, out, gamma_mo);
}
